// OrbEncoder_15693810499873
// MI455X (gfx1250) — matrix operand order repaired and checked by reading, not compiled, not run on hardware
//
#include <hip/hip_runtime.h>
#include <math.h>

// ---------------------------------------------------------------------------
// OrbEncoder for MI455X (gfx1250).
// edge MLP: f16 WMMA (v_wmma_f32_16x16x32_f16) with all weights resident in
// the 320KB WGP LDS, pre-swizzled into native A/B fragment layouts so the
// GEMM inner loop is ds_load_b128 + v_wmma only. A-fragments are hoisted into
// registers (wave's mT is constant), transcendental/rcp paths use single-
// instruction GCN builtins. Featurization + layernorms stay in f32.
// ---------------------------------------------------------------------------

typedef __attribute__((ext_vector_type(16))) _Float16 v16h;
typedef __attribute__((ext_vector_type(8)))  float    v8f;

#define C_DIM     128
#define H_DIM     256
#define E_DIM     72      // RBF(8) * SH(9)
#define K1_PAD    96      // 72 padded to multiple of 32
#define M_TILE    32      // edges per workgroup tile
#define CUTOFF_F  6.0f
#define EPS_F     1e-9f

// LDS partition (halves)
#define W1B_HALVES (16*3*512)   // 16 nTiles x 3 kSlices x 1KB frags  (48 KB)
#define W2B_HALVES (16*8*512)   // (128 KB)
#define W3B_HALVES (8*8*512)    // (64 KB)
#define A1_HALVES  (2*3*512)    // edge-feature tile, A-fragment layout (6 KB)
#define H1_HALVES  (2*8*512)    // (16 KB)
#define H2_HALVES  (2*8*512)    // (16 KB)
#define SMEM_BYTES ((W1B_HALVES+W2B_HALVES+W3B_HALVES+A1_HALVES+H1_HALVES+H2_HALVES)*2 \
                    + (256+256+128+M_TILE*C_DIM)*4)   // = 303,616 B < 320 KB

// --- WMMA B-fragment swizzle (16-bit, 32x16 KxN tile, wave32; ISA 7.12.2) ---
__device__ __forceinline__ int b_swz(int k, int n, int nslices) {
  int s = k >> 5, kin = k & 31;
  int f = (n >> 4) * nslices + s;
  int lane = (n & 15) + (((kin >> 3) & 1) << 4);
  return f * 512 + lane * 16 + ((kin >> 4) << 3) + (kin & 7);
}
// --- WMMA A-fragment swizzle (16-bit, 16x32 MxK tile, wave32; ISA 7.12.2) ---
__device__ __forceinline__ int a_swz(int m, int k, int nslices) {
  int s = k >> 5, kin = k & 31;
  int f = (m >> 4) * nslices + s;
  int g = (kin >> 3) & 1;
  int lane = (m & 15) + (g << 4);
  int v = ((kin >> 4) << 2) + ((kin & 7) >> 1);
  int h = (v << 1) + (kin & 1);
  return f * 512 + lane * 16 + h;
}
__device__ __forceinline__ v16h lds_frag(const _Float16* base, int frag, int lane) {
  return *(const v16h*)(base + frag * 512 + lane * 16);   // 32B contiguous/lane
}
// silu via single v_rcp_f32 instead of IEEE division chain
__device__ __forceinline__ float silu_f(float x) {
  return x * __builtin_amdgcn_rcpf(1.0f + __expf(-x));
}
#define WMMA16(A, B, ACC) \
  __builtin_amdgcn_wmma_f32_16x16x32_f16(false, (A), false, (B), (short)0, (ACC), false, false)

__global__ void __launch_bounds__(256, 1)
edge_mlp_kernel(const float* __restrict__ positions,
                const int*   __restrict__ senders,
                const int*   __restrict__ receivers,
                const float* __restrict__ W1, const float* __restrict__ b1,
                const float* __restrict__ W2, const float* __restrict__ b2,
                const float* __restrict__ W3, const float* __restrict__ b3,
                const float* __restrict__ eln_w, const float* __restrict__ eln_b,
                float* __restrict__ out_edge, int E, int numTiles)
{
  extern __shared__ char smem_raw[];
  _Float16* w1b = (_Float16*)smem_raw;
  _Float16* w2b = w1b + W1B_HALVES;
  _Float16* w3b = w2b + W2B_HALVES;
  _Float16* a1b = w3b + W3B_HALVES;
  _Float16* h1  = a1b + A1_HALVES;
  _Float16* h2  = h1  + H1_HALVES;
  float* b1s = (float*)(h2 + H2_HALVES);
  float* b2s = b1s + 256;
  float* b3s = b2s + 256;
  float* h3  = b3s + 128;                    // M_TILE x 128 f32

  const int tid  = threadIdx.x;
  const int wave = tid >> 5;
  const int lane = tid & 31;

  // ---------- one-time weight staging: f32 -> f16, swizzled B fragments ----
  for (int idx = tid; idx < K1_PAD * H_DIM; idx += 256) {
    int k = idx >> 8, n = idx & 255;
    float v = (k < E_DIM) ? W1[k * H_DIM + n] : 0.0f;     // zero-pad K 72..95
    w1b[b_swz(k, n, 3)] = (_Float16)v;
  }
  for (int idx = tid; idx < H_DIM * H_DIM; idx += 256) {
    int k = idx >> 8, n = idx & 255;
    w2b[b_swz(k, n, 8)] = (_Float16)W2[idx];
  }
  for (int idx = tid; idx < H_DIM * C_DIM; idx += 256) {
    int k = idx >> 7, n = idx & 127;
    w3b[b_swz(k, n, 8)] = (_Float16)W3[idx];
  }
  b1s[tid] = b1[tid];
  b2s[tid] = b2[tid];
  if (tid < 128) b3s[tid] = b3[tid];
  __syncthreads();

  const float s3 = sqrtf(3.0f), s5 = sqrtf(5.0f), s15 = sqrtf(15.0f);
  const float pref = sqrtf(2.0f / CUTOFF_F);
  const float PIf  = 3.14159265358979f;

  // wave-constant tile-role precomputation
  const int mT   = wave & 1;                 // fixed output m-tile per wave
  const int nTb  = wave >> 1;                // n-tile base (stride 4)
  const int mhi  = (lane >> 4) << 3;         // D layout: row = rr + mhi
  // loop-invariant LN params per lane
  const float ew0 = eln_w[lane],    ew1 = eln_w[lane+32];
  const float ew2 = eln_w[lane+64], ew3 = eln_w[lane+96];
  const float eb0 = eln_b[lane],    eb1 = eln_b[lane+32];
  const float eb2 = eln_b[lane+64], eb3 = eln_b[lane+96];
  // per-lane swizzled-store geometry for a given output column n:
  // offset = fragBase + (mhi + g*16)*16 + hh + rr*16
  const int nlo = lane & 15;

  for (int tile = blockIdx.x; tile < numTiles; tile += gridDim.x) {
    const int tileBase = tile * M_TILE;

    // ---------- Phase 1: featurization, 8 threads / edge (one per RBF) ----
    {
      int m = tid >> 3;                 // edge row in tile 0..31
      int r = tid & 7;                  // RBF index
      int e = tileBase + m;
      float vx = 0.f, vy = 0.f, vz = 0.f;
      if (e < E) {
        int si = senders[e], ri = receivers[e];
        vx = positions[3*ri+0] - positions[3*si+0];
        vy = positions[3*ri+1] - positions[3*si+1];
        vz = positions[3*ri+2] - positions[3*si+2];
      }
      float d   = __builtin_amdgcn_sqrtf(vx*vx + vy*vy + vz*vz);
      float inv = __builtin_amdgcn_rcpf(d + EPS_F);
      float x = vx*inv, y = vy*inv, z = vz*inv;
      float sh[9];
      sh[0] = 1.0f;       sh[1] = s3*x;       sh[2] = s3*y;      sh[3] = s3*z;
      sh[4] = s15*x*y;    sh[5] = s15*y*z;    sh[6] = 0.5f*s5*(3.0f*z*z-1.0f);
      sh[7] = s15*x*z;    sh[8] = 0.5f*s15*(x*x - y*y);
      float u  = d * (1.0f / CUTOFF_F);
      float u4 = (u*u)*(u*u);
      float env = (u < 1.0f) ? (1.0f - 15.0f*u4 + 24.0f*u4*u - 10.0f*u4*u*u) : 0.0f;
      float rbf = pref * __sinf((float)(r + 1) * PIf * d * (1.0f / CUTOFF_F)) * inv;
      float scale = (e < E) ? rbf * env : 0.0f;
      #pragma unroll
      for (int q = 0; q < 9; ++q)
        a1b[a_swz(m, r*9 + q, 3)] = (_Float16)(scale * sh[q]);
      #pragma unroll
      for (int j = 0; j < 3; ++j)                        // zero K pad 72..95
        a1b[a_swz(m, E_DIM + r*3 + j, 3)] = (_Float16)0.0f;
    }
    __syncthreads();

    // ---------- GEMM1: [32x96] x [96x256] -> silu -> h1 (A-frag layout) ----
    {
      v16h af0 = lds_frag(a1b, mT*3 + 0, lane);
      v16h af1 = lds_frag(a1b, mT*3 + 1, lane);
      v16h af2 = lds_frag(a1b, mT*3 + 2, lane);
      #pragma unroll
      for (int j = 0; j < 4; ++j) {
        int nT = nTb + j*4;
        v8f acc = {};
        acc = WMMA16(af0, lds_frag(w1b, nT*3 + 0, lane), acc);
        acc = WMMA16(af1, lds_frag(w1b, nT*3 + 1, lane), acc);
        acc = WMMA16(af2, lds_frag(w1b, nT*3 + 2, lane), acc);
        int n   = nT*16 + nlo;
        int kin = n & 31;
        int g   = (kin >> 3) & 1;
        int hh  = ((((kin >> 4) << 2) + ((kin & 7) >> 1)) << 1) + (kin & 1);
        _Float16* dst = h1 + (mT*8 + (n >> 5))*512 + (mhi + (g << 4))*16 + hh;
        float bias = b1s[n];
        #pragma unroll
        for (int rr = 0; rr < 8; ++rr)
          dst[rr*16] = (_Float16)silu_f(acc[rr] + bias);
      }
    }
    __syncthreads();

    // ---------- GEMM2: [32x256] x [256x256] -> silu -> h2 -----------------
    {
      v16h af[8];
      #pragma unroll
      for (int s = 0; s < 8; ++s) af[s] = lds_frag(h1, mT*8 + s, lane);
      #pragma unroll
      for (int j = 0; j < 4; ++j) {
        int nT = nTb + j*4;
        v8f acc = {};
        #pragma unroll
        for (int s = 0; s < 8; ++s)
          acc = WMMA16(af[s], lds_frag(w2b, nT*8 + s, lane), acc);
        int n   = nT*16 + nlo;
        int kin = n & 31;
        int g   = (kin >> 3) & 1;
        int hh  = ((((kin >> 4) << 2) + ((kin & 7) >> 1)) << 1) + (kin & 1);
        _Float16* dst = h2 + (mT*8 + (n >> 5))*512 + (mhi + (g << 4))*16 + hh;
        float bias = b2s[n];
        #pragma unroll
        for (int rr = 0; rr < 8; ++rr)
          dst[rr*16] = (_Float16)silu_f(acc[rr] + bias);
      }
    }
    __syncthreads();

    // ---------- GEMM3: [32x256] x [256x128] + b3 -> h3 (f32) --------------
    {
      v16h af[8];
      #pragma unroll
      for (int s = 0; s < 8; ++s) af[s] = lds_frag(h2, mT*8 + s, lane);
      #pragma unroll
      for (int j = 0; j < 2; ++j) {
        int nT = nTb + j*4;                              // 0..7
        v8f acc = {};
        #pragma unroll
        for (int s = 0; s < 8; ++s)
          acc = WMMA16(af[s], lds_frag(w3b, nT*8 + s, lane), acc);
        int n = nT*16 + nlo;
        float bias = b3s[n];
        float* dst = h3 + (mT*16 + mhi)*C_DIM + n;
        #pragma unroll
        for (int rr = 0; rr < 8; ++rr)
          dst[rr*C_DIM] = acc[rr] + bias;
      }
    }
    __syncthreads();

    // ---------- LayerNorm(128) + store: 1 wave handles 4 rows -------------
    #pragma unroll
    for (int ri = 0; ri < 4; ++ri) {
      int row = wave * 4 + ri;
      int e = tileBase + row;
      float v0 = h3[row*C_DIM + lane];
      float v1 = h3[row*C_DIM + lane + 32];
      float v2 = h3[row*C_DIM + lane + 64];
      float v3 = h3[row*C_DIM + lane + 96];
      float sm = v0+v1+v2+v3;
      float sq = v0*v0+v1*v1+v2*v2+v3*v3;
      #pragma unroll
      for (int off = 16; off > 0; off >>= 1) {           // wave32 reduction
        sm += __shfl_xor(sm, off, 32);
        sq += __shfl_xor(sq, off, 32);
      }
      float mean = sm * (1.0f/128.0f);
      float var  = sq * (1.0f/128.0f) - mean*mean;
      float rstd = __builtin_amdgcn_rsqf(var + 1e-5f);
      if (e < E) {
        float* o = out_edge + (size_t)e * C_DIM;
        o[lane]      = (v0-mean)*rstd*ew0 + eb0;
        o[lane+32]   = (v1-mean)*rstd*ew1 + eb1;
        o[lane+64]   = (v2-mean)*rstd*ew2 + eb2;
        o[lane+96]   = (v3-mean)*rstd*ew3 + eb3;
      }
    }
    __syncthreads();                                     // LDS reuse next tile
  }
}

// ------------------------- node embedding + layernorm ----------------------
__global__ void __launch_bounds__(256)
node_ln_kernel(const int* __restrict__ Z, const float* __restrict__ z_table,
               const float* __restrict__ w, const float* __restrict__ b,
               float* __restrict__ out, int Nn)
{
  int wave = threadIdx.x >> 5, lane = threadIdx.x & 31;
  int node = blockIdx.x * 8 + wave;
  if (node >= Nn) return;
  const float* row = z_table + (size_t)Z[node] * C_DIM;
  float v0 = row[lane], v1 = row[lane+32], v2 = row[lane+64], v3 = row[lane+96];
  float sm = v0+v1+v2+v3;
  float sq = v0*v0+v1*v1+v2*v2+v3*v3;
  #pragma unroll
  for (int off = 16; off > 0; off >>= 1) {
    sm += __shfl_xor(sm, off, 32);
    sq += __shfl_xor(sq, off, 32);
  }
  float mean = sm * (1.0f/128.0f);
  float var  = sq * (1.0f/128.0f) - mean*mean;
  float rstd = __builtin_amdgcn_rsqf(var + 1e-5f);
  float* o = out + (size_t)node * C_DIM;
  o[lane]    = (v0-mean)*rstd*w[lane]    + b[lane];
  o[lane+32] = (v1-mean)*rstd*w[lane+32] + b[lane+32];
  o[lane+64] = (v2-mean)*rstd*w[lane+64] + b[lane+64];
  o[lane+96] = (v3-mean)*rstd*w[lane+96] + b[lane+96];
}

extern "C" void kernel_launch(void* const* d_in, const int* in_sizes, int n_in,
                              void* d_out, int out_size, void* d_ws, size_t ws_size,
                              hipStream_t stream) {
  const int*   Z         = (const int*)  d_in[0];
  const float* positions = (const float*)d_in[1];
  const int*   senders   = (const int*)  d_in[2];
  const int*   receivers = (const int*)  d_in[3];
  const float* z_table   = (const float*)d_in[4];
  const float* zln_w     = (const float*)d_in[5];
  const float* zln_b     = (const float*)d_in[6];
  const float* W1        = (const float*)d_in[7];
  const float* b1        = (const float*)d_in[8];
  const float* W2        = (const float*)d_in[9];
  const float* b2        = (const float*)d_in[10];
  const float* W3        = (const float*)d_in[11];
  const float* b3        = (const float*)d_in[12];
  const float* eln_w     = (const float*)d_in[13];
  const float* eln_b     = (const float*)d_in[14];

  const int Nn = in_sizes[0];
  const int E  = in_sizes[2];
  float* out_node = (float*)d_out;
  float* out_edge = out_node + (size_t)Nn * C_DIM;

  (void)hipFuncSetAttribute((const void*)edge_mlp_kernel,
                            hipFuncAttributeMaxDynamicSharedMemorySize, SMEM_BYTES);

  int numTiles = (E + M_TILE - 1) / M_TILE;
  int grid = numTiles < 2048 ? numTiles : 2048;          // persistent blocks
  edge_mlp_kernel<<<grid, 256, SMEM_BYTES, stream>>>(
      positions, senders, receivers, W1, b1, W2, b2, W3, b3,
      eln_w, eln_b, out_edge, E, numTiles);

  node_ln_kernel<<<(Nn + 7) / 8, 256, 0, stream>>>(Z, z_table, zln_w, zln_b,
                                                   out_node, Nn);
}
